// Model_55886114455791
// MI455X (gfx1250) — matrix operand order repaired and checked by reading, compiled, not run on hardware
//
#include <hip/hip_runtime.h>
#include <hip/hip_bf16.h>

// ---------------------------------------------------------------------------
// Model: windowed rFFT -> top-64 -> complex E-mix -> scatter/irFFT -> overlap
// add -> FC1(46080x512) -> leakyrelu -> FC2(512x96).
// Rank-1-in-E factorization collapses the heavy FC1 (60.6 GFLOP, 237 MB
// activations) into a 1284x5760x512 GEMM (7.6 GFLOP, ~35 MB) against a
// precomputed complex table. All GEMMs run on v_wmma_f32_16x16x32_f16 with
// f32 accumulation, LDS-free: both fragments are loaded straight from global
// memory as contiguous b128 chunks (B matrices are generated transposed).
// ---------------------------------------------------------------------------

typedef __attribute__((ext_vector_type(16))) _Float16 v16h;
typedef __attribute__((ext_vector_type(8)))  _Float16 v8h;
typedef __attribute__((ext_vector_type(8)))  float    v8f;

#define B_    4
#define T_    720
#define N_    321
#define E_    64
#define H_    512
#define P_    96
#define C_    2
#define WS_   480
#define HOP_  240
#define F_    241
#define TOPM_ 64
#define NB_   (B_ * N_)          // 1284 (b,n) rows
#define RC_   (NB_ * C_)         // 2568 (b,n,c) rows
#define KBIG_ (C_ * 3 * WS_ * 2) // 5760 contraction depth

#define TWO_PI 6.28318530717958647692f

// ---------------------------------------------------------------------------
// LDS-free WMMA f16 GEMM: C(MxN,f32) = A(MxK,f16 row-major) * Bt(NxK,f16)^T.
// Block = 128 threads = 4 waves. Block tile = 32 x 64: each wave owns one
// 16-col strip and two 16-row tiles (two independent wmma chains sharing one
// B fragment). Fragments are loaded directly from global in the wave32 VGPR
// layout; out-of-range rows/cols clamp their pointer (their D rows/cols are
// never stored, so no masking is needed). Requires K % 32 == 0, 16B-aligned
// rows (lda/ldbt even multiples of 8 halves).
// ---------------------------------------------------------------------------
__global__ void __launch_bounds__(128)
gemm_f16_wmma(const _Float16* __restrict__ A, int lda,
              const _Float16* __restrict__ Bt, int ldbt,
              float* __restrict__ Cm, int ldc,
              int M, int N, int K) {
    const int tid  = threadIdx.x;
    const int wave = tid >> 5;
    const int lane = tid & 31;
    const int half = lane >> 4;   // 0: lanes 0-15, 1: lanes 16-31
    const int l15  = lane & 15;
    const int tileM = blockIdx.x * 32;
    const int tileN = blockIdx.y * 64 + wave * 16;

    // A fragment rows: lane holds row m=l15; halves split K {0..7,16..23} vs
    // {8..15,24..31}. Two fragments: rows tileM+l15 and tileM+16+l15.
    int arow0 = tileM + l15;
    int arow1 = tileM + 16 + l15;
    const _Float16* ap0 = A + (size_t)((arow0 < M) ? arow0 : 0) * lda + 8 * half;
    const _Float16* ap1 = A + (size_t)((arow1 < M) ? arow1 : 0) * lda + 8 * half;
    // B fragment: lane holds col n=l15; halves split K {0..7,16..23} vs
    // {8..15,24..31}.
    int bcol = tileN + l15;
    const _Float16* bp = Bt + (size_t)((bcol < N) ? bcol : 0) * ldbt + 8 * half;

    v8f accLo = {}, accHi = {};
    for (int k0 = 0; k0 < K; k0 += 32) {
        union { v16h v; v8h h[2]; } a0, a1, b;
        a0.h[0] = *(const v8h*)(ap0 + k0);
        a0.h[1] = *(const v8h*)(ap0 + k0 + 16);
        a1.h[0] = *(const v8h*)(ap1 + k0);
        a1.h[1] = *(const v8h*)(ap1 + k0 + 16);
        b.h[0]  = *(const v8h*)(bp + k0);
        b.h[1]  = *(const v8h*)(bp + k0 + 16);
        if (k0 + 32 < K) { // hint next K-slab (global_prefetch_b8)
            __builtin_prefetch(ap0 + k0 + 32, 0, 1);
            __builtin_prefetch(bp  + k0 + 32, 0, 1);
        }
        accLo = __builtin_amdgcn_wmma_f32_16x16x32_f16(
            false, a0.v, false, b.v, (short)0, accLo, false, false);
        accHi = __builtin_amdgcn_wmma_f32_16x16x32_f16(
            false, a1.v, false, b.v, (short)0, accHi, false, false);
    }
    // D layout: lanes 0-15 rows v (v=0..7), lanes 16-31 rows 8+v; col = l15.
    int n = tileN + l15;
    if (n < N) {
        #pragma unroll
        for (int v = 0; v < 8; ++v) {
            int r0 = tileM + ((lane < 16) ? v : (8 + v));
            int r1 = r0 + 16;
            if (r0 < M) Cm[(size_t)r0 * ldc + n] = accLo[v];
            if (r1 < M) Cm[(size_t)r1 * ldc + n] = accHi[v];
        }
    }
}

// ---------------------------------------------------------------------------
// K1: v0 = emb @ (hc_w[0,0] + i hc_w[0,1]), v1 = emb @ (hc_w[1,0] + i hc_w[1,1])
// ---------------------------------------------------------------------------
__global__ void k_vvec(const float* __restrict__ emb, const float* __restrict__ hcw,
                       float* __restrict__ vbuf) {
    int f = threadIdx.x; // 64 threads
    float s00 = 0.f, s01 = 0.f, s10 = 0.f, s11 = 0.f;
    for (int e = 0; e < E_; ++e) {
        float em = emb[e];
        s00 += em * hcw[((0 * 2 + 0) * E_ + e) * E_ + f];
        s01 += em * hcw[((0 * 2 + 1) * E_ + e) * E_ + f];
        s10 += em * hcw[((1 * 2 + 0) * E_ + e) * E_ + f];
        s11 += em * hcw[((1 * 2 + 1) * E_ + e) * E_ + f];
    }
    vbuf[f] = s00; vbuf[64 + f] = s01; vbuf[128 + f] = s10; vbuf[192 + f] = s11;
}

// K2: transposed rDFT basis (512 x 480 f16): row n<241 = cos(2*pi*n*t/480),
// rows 256..496 = -sin for bin n-256, other rows zero.
__global__ void k_basisT(_Float16* __restrict__ basisT) {
    int gid = blockIdx.x * blockDim.x + threadIdx.x;
    if (gid >= H_ * WS_) return;
    int col = gid / WS_, t = gid - col * WS_;
    float val = 0.f;
    const float w0 = TWO_PI / (float)WS_;
    if (col < F_) {
        int ph = (col * t) % WS_;
        val = cosf(w0 * (float)ph);
    } else if (col >= 256 && col < 256 + F_) {
        int ph = ((col - 256) * t) % WS_;
        val = -sinf(w0 * (float)ph);
    }
    basisT[gid] = (_Float16)val;
}

// K3: window matrix (RC_ x 480 f16): row r=(bn*2+c), Awin[r][t] = x[b, 240c+t, n]
__global__ void k_awin(const float* __restrict__ x, _Float16* __restrict__ AwinH) {
    int gid = blockIdx.x * blockDim.x + threadIdx.x;
    if (gid >= RC_ * WS_) return;
    int r = gid / WS_, t = gid - r * WS_;
    int c = r & 1, bn = r >> 1;
    int b = bn / N_, n = bn - b * N_;
    AwinH[(size_t)r * WS_ + t] = (_Float16)x[((size_t)b * T_ + HOP_ * c + t) * N_ + n];
}

// K4: exact top-64 of |X|^2 over 241 bins (ties -> lower index, jax order).
__global__ void k_topk(const float* __restrict__ spect, int* __restrict__ idxw) {
    int rc = blockIdx.x, tid = threadIdx.x;
    __shared__ float sval[256];
    __shared__ int   sidx[256];
    float my = -1.f; int myf = tid;
    if (tid < F_) {
        float re = spect[(size_t)rc * 512 + tid];
        float im = spect[(size_t)rc * 512 + 256 + tid];
        my = re * re + im * im;
    }
    for (int it = 0; it < TOPM_; ++it) {
        sval[tid] = my; sidx[tid] = myf;
        __syncthreads();
        for (int s = 128; s > 0; s >>= 1) {
            if (tid < s) {
                float v2 = sval[tid + s]; int i2 = sidx[tid + s];
                if (v2 > sval[tid] || (v2 == sval[tid] && i2 < sidx[tid])) {
                    sval[tid] = v2; sidx[tid] = i2;
                }
            }
            __syncthreads();
        }
        int win = sidx[0];
        if (tid == 0) idxw[rc * TOPM_ + it] = win;
        if (myf == win) my = -1.f;
        __syncthreads();
    }
}

// K5: per (b,n,c) complex series S_A,S_B,S_1 over tau in [0,480) -> Sbig rows.
// a_m = X0[idx0[m]], b_m = X1[idx1[m]] (top-k order pairing); phase/coef from
// window c's own bins. Integer mod-480 phase reduction keeps __sincosf exact.
__global__ void k_series(const float* __restrict__ spect, const int* __restrict__ idxw,
                         _Float16* __restrict__ Sbig) {
    int rc = blockIdx.x, tid = threadIdx.x;
    int c = rc & 1, bn = rc >> 1;
    __shared__ float ar[TOPM_], ai[TOPM_], br[TOPM_], bi[TOPM_], co[TOPM_];
    __shared__ int   ks[TOPM_];
    if (tid < TOPM_) {
        int k0 = idxw[(bn * 2 + 0) * TOPM_ + tid];
        int k1 = idxw[(bn * 2 + 1) * TOPM_ + tid];
        int kc = c ? k1 : k0;
        ks[tid] = kc;
        co[tid] = ((kc == 0) || (kc == HOP_)) ? (1.f / WS_) : (2.f / WS_);
        ar[tid] = spect[(size_t)(bn * 2 + 0) * 512 + k0];
        ai[tid] = spect[(size_t)(bn * 2 + 0) * 512 + 256 + k0];
        br[tid] = spect[(size_t)(bn * 2 + 1) * 512 + k1];
        bi[tid] = spect[(size_t)(bn * 2 + 1) * 512 + 256 + k1];
    }
    __syncthreads();
    const float w0 = TWO_PI / (float)WS_;
    for (int tau = tid; tau < WS_; tau += blockDim.x) {
        float SAr = 0.f, SAi = 0.f, SBr = 0.f, SBi = 0.f, S1r = 0.f, S1i = 0.f;
        for (int m = 0; m < TOPM_; ++m) {
            int ph = (ks[m] * tau) % WS_;
            float s, cn;
            __sincosf(w0 * (float)ph, &s, &cn);
            float cf = co[m];
            float er = cf * cn, ei = cf * s;
            SAr += ar[m] * er - ai[m] * ei; SAi += ar[m] * ei + ai[m] * er;
            SBr += br[m] * er - bi[m] * ei; SBi += br[m] * ei + bi[m] * er;
            S1r += er;                       S1i += ei;
        }
        size_t base = (size_t)bn * KBIG_ + (size_t)c * 2880;
        Sbig[base +        2 * tau]     = (_Float16)SAr;
        Sbig[base +        2 * tau + 1] = (_Float16)SAi;
        Sbig[base + 960  + 2 * tau]     = (_Float16)SBr;
        Sbig[base + 960  + 2 * tau + 1] = (_Float16)SBi;
        Sbig[base + 1920 + 2 * tau]     = (_Float16)S1r;
        Sbig[base + 1920 + 2 * tau + 1] = (_Float16)S1i;
    }
}

// K6: transposed Wbig table (512 x 5760 f16) folding u-vectors, overlap-add
// weights and fc_w1. One thread produces the (Re, -Im) pair -> coalesced
// paired stores, half the fc_w1 traffic.
__global__ void k_wbigT(const float* __restrict__ fc_w1, const float* __restrict__ hc_b,
                        const float* __restrict__ vbuf, _Float16* __restrict__ WbigT) {
    int gid = blockIdx.x * blockDim.x + threadIdx.x;
    if (gid >= H_ * (KBIG_ / 2)) return;
    int h = gid / (KBIG_ / 2), kp = gid - h * (KBIG_ / 2);
    int c = kp / 1440, rem = kp - c * 1440;
    int j = rem / 480, tau = rem - j * 480;
    const float* v0r = vbuf;        const float* v0i = vbuf + 64;
    const float* v1r = vbuf + 128;  const float* v1i = vbuf + 192;
    float g = (c == 0) ? ((tau < HOP_) ? 1.f : 0.5f) : ((tau < HOP_) ? 0.5f : 1.f);
    int trow = HOP_ * c + tau; // global time index
    float accR = 0.f, accI = 0.f;
    for (int e = 0; e < E_; ++e) {
        float ur, ui;
        if (j == 0)      { ur = c ? v1r[e] : v0r[e];  ui = c ? v1i[e] : v0i[e]; }   // uA
        else if (j == 1) { ur = c ? v0r[e] : -v1r[e]; ui = c ? -v0i[e] : v1i[e]; }  // uB
        else             { ur = hc_b[(c * 2 + 0) * E_ + e]; ui = hc_b[(c * 2 + 1) * E_ + e]; }
        float w = fc_w1[(size_t)(trow * E_ + e) * H_ + h];
        accR += ur * w; accI += ui * w;
    }
    size_t o = (size_t)h * KBIG_ + 2 * (size_t)kp;
    WbigT[o]     = (_Float16)(g * accR);
    WbigT[o + 1] = (_Float16)(-g * accI);
}

// K7: h = Hpre + b1, leaky relu(0.01), convert to f16 for FC2.
__global__ void k_biaslrelu(const float* __restrict__ Hpre, const float* __restrict__ b1,
                            _Float16* __restrict__ HactH) {
    int gid = blockIdx.x * blockDim.x + threadIdx.x;
    if (gid >= NB_ * H_) return;
    int col = gid & 511;
    float v = Hpre[gid] + b1[col];
    v = (v > 0.f) ? v : 0.01f * v;
    HactH[gid] = (_Float16)v;
}

// K8: fc_w2 -> transposed f16 (96 x 512)
__global__ void k_w2t(const float* __restrict__ w2, _Float16* __restrict__ W2T) {
    int gid = blockIdx.x * blockDim.x + threadIdx.x;
    if (gid >= P_ * H_) return;
    int p = gid / H_, k = gid - p * H_;
    W2T[gid] = (_Float16)w2[(size_t)k * P_ + p];
}

// K9: out[b,p,n] = Opre[bn][p] + b2[p]
__global__ void k_final(const float* __restrict__ Opre, const float* __restrict__ b2,
                        float* __restrict__ out) {
    int gid = blockIdx.x * blockDim.x + threadIdx.x;
    if (gid >= NB_ * P_) return;
    int r = gid / P_, p = gid - r * P_;
    int b = r / N_, n = r - b * N_;
    out[((size_t)b * P_ + p) * N_ + n] = Opre[gid] + b2[p];
}

// ---------------------------------------------------------------------------
extern "C" void kernel_launch(void* const* d_in, const int* in_sizes, int n_in,
                              void* d_out, int out_size, void* d_ws, size_t ws_size,
                              hipStream_t stream) {
    (void)in_sizes; (void)n_in; (void)out_size; (void)ws_size;
    const float* x     = (const float*)d_in[0];
    const float* emb   = (const float*)d_in[1];
    const float* hc_w  = (const float*)d_in[2];
    const float* hc_b  = (const float*)d_in[3];
    const float* fc_w1 = (const float*)d_in[4];
    const float* fc_b1 = (const float*)d_in[5];
    const float* fc_w2 = (const float*)d_in[6];
    const float* fc_b2 = (const float*)d_in[7];
    float* out = (float*)d_out;

    char* ws = (char*)d_ws;
    size_t cur = 0;
    auto take = [&](size_t bytes) -> char* {
        char* p = ws + cur;
        cur = (cur + bytes + 255) & ~(size_t)255;
        return p;
    };
    float*    vbuf   = (float*)   take(4 * E_ * sizeof(float));   // v0,v1 re/im
    _Float16* basisT = (_Float16*)take((size_t)H_ * WS_ * 2);     // DFT basis^T
    _Float16* AwinH  = (_Float16*)take((size_t)RC_ * WS_ * 2);    // windows
    float*    spect  = (float*)   take((size_t)RC_ * 512 * 4);    // spectra
    int*      idxw   = (int*)     take((size_t)RC_ * TOPM_ * 4);  // top-k idx
    _Float16* SbigH  = (_Float16*)take((size_t)NB_ * KBIG_ * 2);  // series
    _Float16* WbigT  = (_Float16*)take((size_t)H_ * KBIG_ * 2);   // folded W1^T
    float*    Hpre   = (float*)   take((size_t)NB_ * H_ * 4);
    _Float16* HactH  = (_Float16*)take((size_t)NB_ * H_ * 2);
    _Float16* W2T    = (_Float16*)take((size_t)P_ * H_ * 2);
    float*    Opre   = (float*)   take((size_t)NB_ * P_ * 4);

    // Precompute (cheap, deterministic each call)
    k_vvec<<<1, 64, 0, stream>>>(emb, hc_w, vbuf);
    k_basisT<<<(H_ * WS_ + 255) / 256, 256, 0, stream>>>(basisT);
    k_w2t<<<(P_ * H_ + 255) / 256, 256, 0, stream>>>(fc_w2, W2T);
    k_awin<<<(RC_ * WS_ + 255) / 256, 256, 0, stream>>>(x, AwinH);

    // GEMM 1: real DFT of all 2568 windows (2568 x 480 x 512)
    gemm_f16_wmma<<<dim3((RC_ + 31) / 32, (512 + 63) / 64), 128, 0, stream>>>(
        AwinH, WS_, basisT, WS_, spect, 512, RC_, 512, WS_);

    k_topk<<<RC_, 256, 0, stream>>>(spect, idxw);
    k_series<<<RC_, 128, 0, stream>>>(spect, idxw, SbigH);
    k_wbigT<<<(H_ * (KBIG_ / 2) + 255) / 256, 256, 0, stream>>>(fc_w1, hc_b, vbuf, WbigT);

    // GEMM 2: the collapsed FC1 (1284 x 5760 x 512)
    gemm_f16_wmma<<<dim3((NB_ + 31) / 32, (H_ + 63) / 64), 128, 0, stream>>>(
        SbigH, KBIG_, WbigT, KBIG_, Hpre, H_, NB_, H_, KBIG_);

    k_biaslrelu<<<(NB_ * H_ + 255) / 256, 256, 0, stream>>>(Hpre, fc_b1, HactH);

    // GEMM 3: FC2 (1284 x 512 x 96)
    gemm_f16_wmma<<<dim3((NB_ + 31) / 32, (P_ + 63) / 64), 128, 0, stream>>>(
        HactH, H_, W2T, H_, Opre, P_, NB_, P_, H_);

    k_final<<<(NB_ * P_ + 255) / 256, 256, 0, stream>>>(Opre, fc_b2, out);
}
